// Prior_83751862272676
// MI455X (gfx1250) — matrix operand order repaired and checked by reading, not compiled, not run on hardware
//
#include <hip/hip_runtime.h>
#include <hip/hip_bf16.h>

// ---------------------------------------------------------------------------
// Problem constants (from reference): N=32768, D=256, M=8192, MU=0.99
// ---------------------------------------------------------------------------
#define NN   32768
#define DD   256
#define MM   8192
#define MU_F 0.99f

typedef __bf16 bf16_t;
typedef __attribute__((ext_vector_type(8)))  bf16_t v8bf;
typedef __attribute__((ext_vector_type(16))) bf16_t v16bf;
typedef __attribute__((ext_vector_type(8)))  float  v8f;

// LDS row stride in bf16 elements: 264*2 = 528 bytes (multiple of 16 -> aligned
// 128-bit ds loads; 528/4 = 132 dwords -> 4-bank rotation per row, conflict-free
// across 16 rows).
#define LSTR 264

// ------------------------- output layout (floats) --------------------------
// prior [M*D] | zi [N] | zq [N*D] | loss [1] | new_sum [M*D] | new_elem [M]
#define O_PRIOR 0
#define O_ZI    (MM * DD)
#define O_ZQ    (O_ZI + NN)
#define O_LOSS  (O_ZQ + (size_t)NN * DD)
#define O_NSUM  (O_LOSS + 1)
#define O_NELEM (O_NSUM + MM * DD)

// ------------------------- workspace layout (bytes) ------------------------
#define W_PRIORBF 0                               // M*D bf16 = 4 MiB
#define W_NORM    (MM * DD * 2)                   // M floats
#define W_SUMCUR  (W_NORM + MM * 4)               // M*D floats
#define W_COUNT   (W_SUMCUR + MM * DD * 4)        // M floats
#define W_LOSS    (W_COUNT + MM * 4)              // 1 float
#define ZERO_FLOATS (MM * DD + MM + 1)            // sum_cur + count + loss

// ---------------------------------------------------------------------------
// K0: zero the accumulation regions of the workspace (deterministic per call)
// ---------------------------------------------------------------------------
__global__ void vq_zero(float* __restrict__ p, int n) {
  int i = blockIdx.x * blockDim.x + threadIdx.x;
  if (i < n) p[i] = 0.0f;
}

// ---------------------------------------------------------------------------
// K1: prior = prior_sum / prior_elem ; also bf16 copy + ||prior||^2 per row
// One 256-thread block per codebook row.
// ---------------------------------------------------------------------------
__global__ __launch_bounds__(256)
void vq_prep(const float* __restrict__ prior_sum,
             const float* __restrict__ prior_elem,
             float*  __restrict__ prior_out,
             bf16_t* __restrict__ prior_bf,
             float*  __restrict__ prior_norm) {
  const int m = blockIdx.x;
  const int c = threadIdx.x;
  const float inv = 1.0f / prior_elem[m];
  const float p = prior_sum[(size_t)m * DD + c] * inv;
  prior_out[(size_t)m * DD + c] = p;
  prior_bf[(size_t)m * DD + c]  = (bf16_t)p;

  __shared__ float red[256];
  red[c] = p * p;
  __syncthreads();
  for (int s = 128; s > 0; s >>= 1) {
    if (c < s) red[c] += red[c + s];
    __syncthreads();
  }
  if (c == 0) prior_norm[m] = red[0];
}

// ---------------------------------------------------------------------------
// K2: fused bf16-WMMA distance + argmin + gather + segment-sum + commit loss.
// 128 threads = 4 waves; block owns 64 ze rows (one 16-row WMMA tile / wave).
// ---------------------------------------------------------------------------
__global__ __launch_bounds__(128)
void vq_argmin(const float*  __restrict__ ze,
               const bf16_t* __restrict__ prior_bf,
               const float*  __restrict__ prior_norm,
               const float*  __restrict__ prior_f32,
               float* __restrict__ zi_out,
               float* __restrict__ zq_out,
               float* __restrict__ sum_cur,
               float* __restrict__ count,
               float* __restrict__ loss_acc) {
  __shared__ __attribute__((aligned(16))) bf16_t ldsA[64 * LSTR]; // ze tile (bf16)
  __shared__ __attribute__((aligned(16))) bf16_t ldsB[16 * LSTR]; // prior tile
  __shared__ int s_idx[4][16];

  const int tid  = threadIdx.x;
  const int lane = tid & 31;
  const int w    = tid >> 5;          // wave id 0..3
  const int n0   = blockIdx.x * 64;   // base ze row for this block

  // ---- stage 64x256 ze tile into LDS as bf16 (each thread 32 float4s) ----
  for (int i = tid; i < (64 * DD) / 4; i += 128) {
    const int row = (i * 4) >> 8;          // /256
    const int col = (i * 4) & (DD - 1);
    const float4 v = *(const float4*)(ze + (size_t)(n0 + row) * DD + col);
    bf16_t* d = ldsA + row * LSTR + col;
    d[0] = (bf16_t)v.x; d[1] = (bf16_t)v.y;
    d[2] = (bf16_t)v.z; d[3] = (bf16_t)v.w;
  }

  // A fragment addressing (16-bit A 16x32 layout):
  //   lanes 0-15: M=lane, K in {k0..k0+7, k0+16..k0+23}
  //   lanes16-31: M=lane-16, K in {k0+8..k0+15, k0+24..k0+31}
  const int rowA = (w << 4) + (lane & 15);
  const int klo  = (lane & 16) ? 8 : 0;
  // B fragment addressing (16-bit B 32x16 layout):
  //   lanes 0-15: N=lane,    K in {k0..k0+7, k0+16..k0+23}
  //   lanes16-31: N=lane-16, K in {k0+8..k0+15, k0+24..k0+31}
  const int rowB = (lane & 15);
  const int kbo  = (lane & 16) ? 8 : 0;

  float bestVal[8];
  int   bestIdx[8];
#pragma unroll
  for (int r = 0; r < 8; ++r) { bestVal[r] = 3.4e38f; bestIdx[r] = 0; }

  for (int mt = 0; mt < MM / 16; ++mt) {
    // ---- cooperative stage of 16x256 bf16 prior tile (512 x 16B chunks) ----
    const int m0 = mt * 16;
    for (int i = tid; i < 512; i += 128) {
      const int row = i >> 5;             // 32 chunks per row
      const int col = (i & 31) * 8;
      *(v8bf*)(ldsB + row * LSTR + col) =
          *(const v8bf*)(prior_bf + (size_t)(m0 + row) * DD + col);
    }
    __syncthreads();   // tiles ready (covers ldsA on first iteration too)

    const float normv = prior_norm[m0 + (lane & 15)];

    v8f c = {0.f, 0.f, 0.f, 0.f, 0.f, 0.f, 0.f, 0.f};
#pragma unroll
    for (int k0 = 0; k0 < DD; k0 += 32) {
      const v8bf a0 = *(const v8bf*)(ldsA + rowA * LSTR + k0 + klo);
      const v8bf a1 = *(const v8bf*)(ldsA + rowA * LSTR + k0 + 16 + klo);
      const v8bf b0 = *(const v8bf*)(ldsB + rowB * LSTR + k0 + kbo);
      const v8bf b1 = *(const v8bf*)(ldsB + rowB * LSTR + k0 + 16 + kbo);
      const v16bf a = __builtin_shufflevector(a0, a1, 0,1,2,3,4,5,6,7,
                                                      8,9,10,11,12,13,14,15);
      const v16bf b = __builtin_shufflevector(b0, b1, 0,1,2,3,4,5,6,7,
                                                      8,9,10,11,12,13,14,15);
      c = __builtin_amdgcn_wmma_f32_16x16x32_bf16(
              /*neg_a=*/false, a, /*neg_b=*/false, b,
              /*c_mod=*/(short)0, c, /*reuse_a=*/false, /*reuse_b=*/false);
    }

    // distance (minus constant ||ze||^2 term): ||e||^2 - 2*dot
    const int mcol = m0 + (lane & 15);
#pragma unroll
    for (int r = 0; r < 8; ++r) {
      const float d = fmaf(-2.0f, c[r], normv);
      if (d < bestVal[r]) { bestVal[r] = d; bestIdx[r] = mcol; }
    }
    __syncthreads();   // done reading ldsB before next stage overwrites it
  }

  // ---- argmin reduction across the 16 lanes holding the same rows ----
  // C layout: VGPR r -> row r (lanes 0-15) / row 8+r (lanes 16-31); col = lane&15
#pragma unroll
  for (int r = 0; r < 8; ++r) {
#pragma unroll
    for (int off = 8; off >= 1; off >>= 1) {
      const float ov = __shfl_xor(bestVal[r], off);
      const int   oi = __shfl_xor(bestIdx[r], off);
      if (ov < bestVal[r] || (ov == bestVal[r] && oi < bestIdx[r])) {
        bestVal[r] = ov; bestIdx[r] = oi;
      }
    }
  }

  if ((lane & 15) == 0) {
    const int half = lane >> 4;   // 0: rows 0-7, 1: rows 8-15
#pragma unroll
    for (int r = 0; r < 8; ++r) {
      const int rowLocal = half * 8 + r;
      s_idx[w][rowLocal] = bestIdx[r];
      zi_out[n0 + w * 16 + rowLocal] = (float)bestIdx[r];
    }
  }
  __syncthreads();

  // ---- gather zq, commit-loss partial, segment-sum scatter ----
  float lsum = 0.0f;
  for (int row = 0; row < 16; ++row) {
    const int idx = s_idx[w][row];
    const size_t n = (size_t)(n0 + w * 16 + row);
    const float4* pp = (const float4*)(prior_f32 + (size_t)idx * DD);
    const float4* zp = (const float4*)(ze + n * DD);
    float4* qp = (float4*)(zq_out + n * DD);
    float* sc = sum_cur + (size_t)idx * DD;
#pragma unroll
    for (int j = 0; j < 2; ++j) {
      const int c4 = lane * 2 + j;        // 0..63 float4 chunks
      const float4 pv = pp[c4];
      const float4 zv = zp[c4];
      qp[c4] = pv;
      const float dx = zv.x - pv.x, dy = zv.y - pv.y;
      const float dz = zv.z - pv.z, dw = zv.w - pv.w;
      lsum += dx * dx + dy * dy + dz * dz + dw * dw;
      atomicAdd(sc + c4 * 4 + 0, zv.x);
      atomicAdd(sc + c4 * 4 + 1, zv.y);
      atomicAdd(sc + c4 * 4 + 2, zv.z);
      atomicAdd(sc + c4 * 4 + 3, zv.w);
    }
    if (lane == 0) atomicAdd(count + idx, 1.0f);
  }
#pragma unroll
  for (int off = 16; off >= 1; off >>= 1) lsum += __shfl_xor(lsum, off);
  if (lane == 0) atomicAdd(loss_acc, lsum);
}

// ---------------------------------------------------------------------------
// K3: EMA update + final loss scale
// ---------------------------------------------------------------------------
__global__ void vq_finalize(const float* __restrict__ prior_sum,
                            const float* __restrict__ prior_elem,
                            const float* __restrict__ sum_cur,
                            const float* __restrict__ count,
                            const float* __restrict__ loss_acc,
                            float* __restrict__ new_sum,
                            float* __restrict__ new_elem,
                            float* __restrict__ loss_out) {
  const int i = blockIdx.x * blockDim.x + threadIdx.x;
  if (i < MM * DD) new_sum[i] = MU_F * prior_sum[i] + (1.0f - MU_F) * sum_cur[i];
  if (i < MM)      new_elem[i] = MU_F * prior_elem[i] + (1.0f - MU_F) * count[i];
  if (i == 0)      *loss_out = *loss_acc * (1.0f / ((float)NN * (float)DD));
}

// ---------------------------------------------------------------------------
extern "C" void kernel_launch(void* const* d_in, const int* in_sizes, int n_in,
                              void* d_out, int out_size, void* d_ws, size_t ws_size,
                              hipStream_t stream) {
  (void)in_sizes; (void)n_in; (void)out_size; (void)ws_size;

  const float* ze         = (const float*)d_in[0];   // [N, D]
  const float* prior_sum  = (const float*)d_in[1];   // [M, D]
  const float* prior_elem = (const float*)d_in[2];   // [M]

  float* out = (float*)d_out;
  float* o_prior = out + O_PRIOR;
  float* o_zi    = out + O_ZI;
  float* o_zq    = out + O_ZQ;
  float* o_loss  = out + O_LOSS;
  float* o_nsum  = out + O_NSUM;
  float* o_nelem = out + O_NELEM;

  char* ws = (char*)d_ws;
  bf16_t* w_priorbf = (bf16_t*)(ws + W_PRIORBF);
  float*  w_norm    = (float*)(ws + W_NORM);
  float*  w_sumcur  = (float*)(ws + W_SUMCUR);
  float*  w_count   = (float*)(ws + W_COUNT);
  float*  w_loss    = (float*)(ws + W_LOSS);

  // K0: zero accumulators (sum_cur, count, loss) — contiguous region
  {
    const int n = ZERO_FLOATS;
    vq_zero<<<(n + 255) / 256, 256, 0, stream>>>(w_sumcur, n);
  }

  // K1: centroids + bf16 copy + norms
  vq_prep<<<MM, 256, 0, stream>>>(prior_sum, prior_elem,
                                  o_prior, w_priorbf, w_norm);

  // K2: WMMA distance + argmin + gather + scatter (512 blocks x 4 waves)
  vq_argmin<<<NN / 64, 128, 0, stream>>>(ze, w_priorbf, w_norm, o_prior,
                                         o_zi, o_zq, w_sumcur, w_count, w_loss);

  // K3: EMA + loss
  {
    const int n = MM * DD;
    vq_finalize<<<(n + 255) / 256, 256, 0, stream>>>(
        prior_sum, prior_elem, w_sumcur, w_count, w_loss,
        o_nsum, o_nelem, o_loss);
  }
}
